// InteractionBlock_79886391705997
// MI455X (gfx1250) — matrix operand order repaired and checked by reading, not compiled, not run on hardware
//
#include <hip/hip_runtime.h>
#include <hip/hip_bf16.h>

typedef __attribute__((ext_vector_type(16))) _Float16 v16h;
typedef __attribute__((ext_vector_type(8)))  float    v8f;

#define NNODE 4096
#define NEDGE 32768
#define FDIM  128
#define NBAS  32

// ---------------------------------------------------------------------------
// Real Clebsch-Gordan sparse tables (ORDER=2, 15 paths), fully constexpr so
// the path loop unrolls into straight-line VALU code (no scratch).
// Path order matches reference PATHS generation.
// ---------------------------------------------------------------------------
static constexpr int CG_L1[15] = {0,0,0,1,1,1,1,1,1,2,2,2,2,2,2};
static constexpr int CG_L2[15] = {0,1,2,0,1,1,1,2,2,0,1,1,2,2,2};
static constexpr int CG_L3[15] = {0,1,2,1,0,1,2,1,2,2,1,2,0,1,2};
static constexpr int CG_START[16] = {0,1,4,9,12,15,21,32,43,55,60,71,83,88,98,110};

static constexpr int CG_P[110] = {
  /*P0*/ 0,
  /*P1*/ 0,0,0,
  /*P2*/ 0,0,0,0,0,
  /*P3*/ 0,1,2,
  /*P4*/ 0,1,2,
  /*P5*/ 0,1,2,0,1,2,
  /*P6*/ 0,2,0,1,1,0,2,2,1,2,0,
  /*P7*/ 2,2,0,1,0,0,2,1,1,2,0,
  /*P8*/ 0,0,0,0,0,0,1,1,1,1,2,2,
  /*P9*/ 0,1,2,3,4,
  /*P10*/ 2,4,0,3,2,4,0,1,2,3,1,
  /*P11*/ 0,3,1,2,1,4,0,4,1,3,2,3,
  /*P12*/ 0,1,2,3,4,
  /*P13*/ 0,1,0,3,1,2,3,4,2,3,
  /*P14*/ 0,1,3,4,2,0,1,1,3,0,3,4
};
static constexpr int CG_Q[110] = {
  /*P0*/ 0,
  /*P1*/ 0,1,2,
  /*P2*/ 0,1,2,3,4,
  /*P3*/ 0,0,0,
  /*P4*/ 0,1,2,
  /*P5*/ 1,2,0,2,0,1,
  /*P6*/ 2,0,1,0,1,0,2,1,2,2,0,
  /*P7*/ 2,4,0,3,2,4,0,1,2,3,1,
  /*P8*/ 0,3,1,2,1,4,0,4,1,3,2,3,
  /*P9*/ 0,0,0,0,0,
  /*P10*/ 2,2,0,1,0,0,2,1,1,2,0,
  /*P11*/ 0,0,0,0,0,0,1,1,1,1,2,2,
  /*P12*/ 0,1,2,3,4,
  /*P13*/ 1,0,3,0,2,1,4,3,3,2,
  /*P14*/ 0,1,3,4,2,1,0,3,1,3,0,0
};
static constexpr int CG_R[110] = {
  /*P0*/ 0,
  /*P1*/ 0,1,2,
  /*P2*/ 0,1,2,3,4,
  /*P3*/ 0,1,2,
  /*P4*/ 0,0,0,
  /*P5*/ 2,0,1,1,2,0,
  /*P6*/ 0,0,1,1,2,2,2,3,3,4,4,
  /*P7*/ 2,2,2,2,0,0,0,0,1,1,1,
  /*P8*/ 3,0,2,1,4,1,4,0,3,1,3,2,
  /*P9*/ 0,1,2,3,4,
  /*P10*/ 2,2,2,2,0,0,0,0,1,1,1,
  /*P11*/ 3,0,2,1,4,1,4,0,3,1,3,2,
  /*P12*/ 0,0,0,0,0,
  /*P13*/ 2,2,0,0,1,1,1,1,0,0,
  /*P14*/ 2,2,2,2,2,3,3,0,0,1,1,0
};
static constexpr float CG_C[110] = {
  /*P0*/ 1.0f,
  /*P1*/ 1.0f,1.0f,1.0f,
  /*P2*/ 1.0f,1.0f,1.0f,1.0f,1.0f,
  /*P3*/ 1.0f,1.0f,1.0f,
  /*P4*/ 0.57735027f,0.57735027f,0.57735027f,
  /*P5*/ 0.70710678f,0.70710678f,0.70710678f,-0.70710678f,-0.70710678f,-0.70710678f,
  /*P6*/ 0.70710678f,0.70710678f,0.70710678f,0.70710678f,0.81649658f,-0.40824829f,
         -0.40824829f,0.70710678f,0.70710678f,0.70710678f,-0.70710678f,
  /*P7*/ -0.31622777f,0.54772256f,0.54772256f,0.54772256f,-0.31622777f,-0.54772256f,
         0.54772256f,0.54772256f,0.63245553f,0.54772256f,0.54772256f,
  /*P8*/ 0.54772256f,-0.54772256f,0.94868330f,-0.94868330f,0.54772256f,-0.54772256f,
         0.54772256f,-0.54772256f,0.54772256f,-0.54772256f,0.94868330f,-0.94868330f,
  /*P9*/ 1.0f,1.0f,1.0f,1.0f,1.0f,
  /*P10*/ -0.31622777f,0.54772256f,0.54772256f,0.54772256f,-0.31622777f,-0.54772256f,
          0.54772256f,0.54772256f,0.63245553f,0.54772256f,0.54772256f,
  /*P11*/ 0.54772256f,-0.54772256f,0.94868330f,-0.94868330f,0.54772256f,-0.54772256f,
          0.54772256f,-0.54772256f,0.54772256f,-0.54772256f,0.94868330f,-0.94868330f,
  /*P12*/ 0.44721360f,0.44721360f,0.44721360f,0.44721360f,0.44721360f,
  /*P13*/ 0.54772256f,-0.54772256f,0.54772256f,-0.54772256f,0.63245553f,-0.63245553f,
          0.54772256f,-0.54772256f,0.63245553f,-0.63245553f,
  /*P14*/ -0.26726124f,0.26726124f,0.26726124f,-0.53452248f,-0.53452248f,0.46291005f,
          0.46291005f,0.46291005f,0.46291005f,0.46291005f,0.46291005f,0.46291005f
};

// ---------------------------------------------------------------------------
// fp32 -> fp16 weight pre-conversion (one-time per launch; weights reused by
// hundreds of blocks afterwards at half the bandwidth, zero per-tile cvt).
// ---------------------------------------------------------------------------
__global__ __launch_bounds__(256) void cvt_f16_kernel(
    const float* __restrict__ src, _Float16* __restrict__ dst, int n) {
  int i = blockIdx.x * 256 + threadIdx.x;
  if (i < n) dst[i] = (_Float16)src[i];
}

// ---------------------------------------------------------------------------
// Swish: y[n,f] = alpha[f] * x * sigmoid(beta[f]*x)
// ---------------------------------------------------------------------------
__global__ __launch_bounds__(256) void swish_kernel(
    const float* __restrict__ x, const float* __restrict__ alpha,
    const float* __restrict__ beta, float* __restrict__ y, int rows) {
  int i = blockIdx.x * 256 + threadIdx.x;
  if (i < rows * FDIM) {
    int f = i & (FDIM - 1);
    float v = x[i];
    float t = beta[f] * v;
    y[i] = alpha[f] * v * (1.0f / (1.0f + __expf(-t)));
  }
}

// ---------------------------------------------------------------------------
// WMMA GEMM: Y[M,128] = X[M,128] @ W[128,128]^T (+bias[g]) (+add[row*128+g])
// W pre-converted to f16. Block = 256 threads (8 waves), 64-row macro-tile:
// each wave owns one 16-col tile across 4 row strips (4 accumulators), so the
// B fragment is loaded once per K-step and reused 4x. 16 chained
// v_wmma_f32_16x16x32_f16 per wave. A strip staged in LDS as f16 (one cvt per
// element). Output row remap folds the final channel-concat:
//   o = (r/mcount)*pitch + (moff + r%mcount)*128 + g
// ---------------------------------------------------------------------------
__global__ __launch_bounds__(256) void gemm_wmma_kernel(
    const float* __restrict__ X, const _Float16* __restrict__ W,
    const float* __restrict__ bias, const float* __restrict__ add,
    float* __restrict__ Y, int M, int mcount, int moff, int pitch) {
  __shared__ _Float16 Xs[64][FDIM];   // 16 KB
  const int rowBase = blockIdx.x * 64;
  const int tid = threadIdx.x;

  // Prefetch the weight panel into GL2 (global_prefetch_b8); 256 threads x
  // 128B lines cover the full 32 KB f16 panel.
  __builtin_prefetch(&W[(size_t)tid * 64], 0, 1);

  for (int i = tid; i < 64 * FDIM; i += 256) {
    int r = i >> 7, c = i & 127;
    Xs[r][c] = (_Float16)X[(size_t)(rowBase + r) * FDIM + c];
  }
  __syncthreads();

  const int wave = tid >> 5;
  const int lane = tid & 31;
  const int col0 = wave * 16;
  const int lrow = lane & 15;    // A: M index within strip; B: N index
  const bool hi = lane >= 16;

  v8f acc[4] = {{}, {}, {}, {}};
#pragma unroll
  for (int kt = 0; kt < 4; ++kt) {
    const int k = kt * 32;
    // B 32x16 f16, this code's contiguous k order (matches A): lane = N, lanes 0..15 hold K=0..15, 16..31 hold 16..31
    v16h b;
    const _Float16* wrow = &W[(size_t)(col0 + lrow) * FDIM + k + (hi ? 16 : 0)];
#pragma unroll
    for (int j = 0; j < 16; ++j) b[j] = wrow[j];
#pragma unroll
    for (int s = 0; s < 4; ++s) {
      // A 16x32 f16, this code's contiguous k order (matches B): lanes<16 hold K=0..15, lanes>=16 hold K=16..31
      v16h a;
      const _Float16* xrow = &Xs[s * 16 + lrow][k + (hi ? 16 : 0)];
#pragma unroll
      for (int j = 0; j < 8; ++j)  a[j] = xrow[j];
#pragma unroll
      for (int j = 8; j < 16; ++j) a[j] = xrow[j];
      acc[s] = __builtin_amdgcn_wmma_f32_16x16x32_f16(false, a, false, b, (short)0,
                                                      acc[s], false, false);
    }
  }
#pragma unroll
  for (int s = 0; s < 4; ++s) {
#pragma unroll
    for (int d = 0; d < 8; ++d) {
      int r = rowBase + s * 16 + d + (hi ? 8 : 0);  // D: VGPR d -> M=d / d+8
      int c = col0 + lrow;
      float v = acc[s][d];
      if (bias) v += bias[c];
      if (add)  v += add[(size_t)r * FDIM + c];
      size_t o = (size_t)(r / mcount) * pitch + (size_t)(moff + (r % mcount)) * FDIM + c;
      Y[o] = v;
    }
  }
}

// ---------------------------------------------------------------------------
// Edge kernel: per 16-edge tile
//   Phase 1 (WMMA): R[path][e][f] = rbf[e,:] @ mixw[path][f,:]  (15 paths,
//                   f16 weights pre-converted, result cached f16 in LDS)
//   Phase 2 (VALU): CG tensor product (fully unrolled constexpr table),
//                   radial gate, atomic scatter-add into v*.
// ---------------------------------------------------------------------------
struct EdgeParams {
  const float* rbf;
  const float* sph0; const float* sph1; const float* sph2;
  const int* idx_i;  const int* idx_j;
  const float* yj0;  const float* yj1;  const float* yj2;
  const _Float16* mixh[15];
  const float* radw[3];
  const float* a1w0; const float* a1w1; const float* a1w2; const float* a1b;
  const float* a2w0; const float* a2w1; const float* a2w2; const float* a2b;
  float* v0; float* v1; float* v2;
};

__global__ __launch_bounds__(128) void edge_kernel(EdgeParams P) {
  __shared__ float    rbf_s[16][NBAS];          // 2 KB
  __shared__ _Float16 R_s[15][16][FDIM];        // 60 KB
  __shared__ float    sph_s[16][9];
  __shared__ int      ii_s[16], ij_s[16];

  const int eb = blockIdx.x * 16;
  const int tid = threadIdx.x;

  for (int t = tid; t < 16 * NBAS; t += 128)
    rbf_s[t >> 5][t & 31] = P.rbf[(size_t)(eb + (t >> 5)) * NBAS + (t & 31)];
  for (int t = tid; t < 16 * 9; t += 128) {
    int e = t / 9, c = t % 9;
    float v;
    if (c == 0)      v = P.sph0[eb + e];
    else if (c < 4)  v = P.sph1[(size_t)(eb + e) * 3 + (c - 1)];
    else             v = P.sph2[(size_t)(eb + e) * 5 + (c - 4)];
    sph_s[e][c] = v;
  }
  if (tid < 16) { ii_s[tid] = P.idx_i[eb + tid]; ij_s[tid] = P.idx_j[eb + tid]; }
  __syncthreads();

  // ---- Phase 1: radial gates via WMMA (K=32 -> exactly one wmma per tile) --
  const int wave = tid >> 5, lane = tid & 31;
  const int lrow = lane & 15;
  const bool hi = lane >= 16;
  v16h a;
#pragma unroll
  for (int j = 0; j < 16; ++j) {
    int ka = (hi ? 16 : 0) + j;
    a[j] = (_Float16)rbf_s[lrow][ka];
  }
  for (int p = wave; p < 15; p += 4) {
    const _Float16* Wp = P.mixh[p];
    for (int ct = 0; ct < 8; ++ct) {
      int col0 = ct * 16;
      v16h b;
      const _Float16* wrow = &Wp[(size_t)(col0 + lrow) * NBAS + (hi ? 16 : 0)];
#pragma unroll
      for (int j = 0; j < 16; ++j) b[j] = wrow[j];
      v8f c = {};
      c = __builtin_amdgcn_wmma_f32_16x16x32_f16(false, a, false, b, (short)0, c,
                                                 false, false);
#pragma unroll
      for (int d = 0; d < 8; ++d)
        R_s[p][d + (hi ? 8 : 0)][col0 + lrow] = (_Float16)c[d];
    }
  }
  __syncthreads();

  // ---- Phase 2: CG mixing + radial message + atomic scatter ----
  const int f = tid;
  const float w1v0 = P.a1w0[f], w1v1 = P.a1w1[f], w1v2 = P.a1w2[f], b1v = P.a1b[f];
  const float w2v0 = P.a2w0[f], w2v1 = P.a2w1[f], w2v2 = P.a2w2[f], b2v = P.a2b[f];

  for (int e = 0; e < 16; ++e) {
    const int jn = ij_s[e], in = ii_s[e];
    const float yj0v = P.yj0[(size_t)jn * FDIM + f];
    float yj1v[3], yj2v[5];
#pragma unroll
    for (int p = 0; p < 3; ++p) yj1v[p] = P.yj1[((size_t)jn * 3 + p) * FDIM + f];
#pragma unroll
    for (int p = 0; p < 5; ++p) yj2v[p] = P.yj2[((size_t)jn * 5 + p) * FDIM + f];

    const float a10 = sph_s[e][0] * w1v0 + b1v;
    float a11[3], a12[5];
#pragma unroll
    for (int q = 0; q < 3; ++q) a11[q] = sph_s[e][1 + q] * w1v1;
#pragma unroll
    for (int q = 0; q < 5; ++q) a12[q] = sph_s[e][4 + q] * w1v2;
    const float a20 = sph_s[e][0] * w2v0 + b2v;
    float a21[3], a22[5];
#pragma unroll
    for (int q = 0; q < 3; ++q) a21[q] = sph_s[e][1 + q] * w2v1;
#pragma unroll
    for (int q = 0; q < 5; ++q) a22[q] = sph_s[e][4 + q] * w2v2;

    float v0a = 0.f, v1a[3] = {0.f, 0.f, 0.f}, v2a[5] = {0.f, 0.f, 0.f, 0.f, 0.f};

#pragma unroll
    for (int p = 0; p < 15; ++p) {
      float t[5] = {0.f, 0.f, 0.f, 0.f, 0.f};
#pragma unroll
      for (int s = CG_START[p]; s < CG_START[p + 1]; ++s) {
        float xa = (CG_L1[p] == 0) ? yj0v
                 : (CG_L1[p] == 1) ? yj1v[CG_P[s]] : yj2v[CG_P[s]];
        float xb = (CG_L2[p] == 0) ? a10
                 : (CG_L2[p] == 1) ? a11[CG_Q[s]] : a12[CG_Q[s]];
        t[CG_R[s]] += CG_C[s] * xa * xb;
      }
      const float Rp = (float)R_s[p][e][f];
      if (CG_L3[p] == 0) {
        v0a += Rp * t[0];
      } else if (CG_L3[p] == 1) {
#pragma unroll
        for (int r = 0; r < 3; ++r) v1a[r] += Rp * t[r];
      } else {
#pragma unroll
        for (int r = 0; r < 5; ++r) v2a[r] += Rp * t[r];
      }
    }

    float rad[3];
#pragma unroll
    for (int L = 0; L < 3; ++L) {
      float s = 0.f;
      const float* rw = P.radw[L] + (size_t)f * NBAS;
#pragma unroll
      for (int n = 0; n < NBAS; ++n) s += rbf_s[e][n] * rw[n];
      rad[L] = s;
    }

    atomicAdd(&P.v0[(size_t)in * FDIM + f], v0a + rad[0] * a20 * yj0v);
#pragma unroll
    for (int m = 0; m < 3; ++m)
      atomicAdd(&P.v1[((size_t)in * 3 + m) * FDIM + f], v1a[m] + rad[1] * a21[m] * yj0v);
#pragma unroll
    for (int m = 0; m < 5; ++m)
      atomicAdd(&P.v2[((size_t)in * 5 + m) * FDIM + f], v2a[m] + rad[2] * a22[m] * yj0v);
  }
}

// ---------------------------------------------------------------------------
// Host-side launch orchestration (all on `stream`; no allocs, graph-safe)
// ---------------------------------------------------------------------------
static inline void launch_gemm(hipStream_t s, const float* X, const _Float16* W,
                               const float* b, const float* add, float* Y, int M,
                               int mc, int mo, int pitch) {
  gemm_wmma_kernel<<<M / 64, 256, 0, s>>>(X, W, b, add, Y, M, mc, mo, pitch);
}
static inline void launch_swish(hipStream_t s, const float* x, const float* al,
                                const float* be, float* y, int rows) {
  swish_kernel<<<(rows * FDIM + 255) / 256, 256, 0, s>>>(x, al, be, y, rows);
}

extern "C" void kernel_launch(void* const* d_in, const int* in_sizes, int n_in,
                              void* d_out, int out_size, void* d_ws, size_t ws_size,
                              hipStream_t stream) {
  (void)in_sizes; (void)n_in; (void)out_size; (void)ws_size;
  const float* x0   = (const float*)d_in[0];
  const float* x1   = (const float*)d_in[1];
  const float* x2   = (const float*)d_in[2];
  const float* rbf  = (const float*)d_in[3];
  const float* sph0 = (const float*)d_in[4];
  const float* sph1 = (const float*)d_in[5];
  const float* sph2 = (const float*)d_in[6];
  const int* idx_i  = (const int*)d_in[7];
  const int* idx_j  = (const int*)d_in[8];
  auto pf = [&](int i) { return (const float*)d_in[i]; };
  // Param leaf order (jax pytree, sorted dict keys), base index 9:
  //  9,10 act_i.{alpha,beta}  11,12 act_j  13,14 act_v
  //  15 ang1.b 16-18 ang1.w[0..2]   19 ang2.b 20-22 ang2.w
  //  23 linear_i.b 24-26 w   27 linear_j.b 28-30 w   31 linear_v.b 32-34 w
  //  35-49 mixing.w[0..14]   50-52 radial[0..2]
  //  53-64 res_post_v (a1a,a1b,a2a,a2b, l1b,l1w0,l1w1,l1w2, l2b,l2w0,l2w1,l2w2)
  //  65-76 res_pre_vi (same)  77-88 res_pre_vj (same)

  float* ws = (float*)d_ws;
  const size_t U = (size_t)NNODE * FDIM;
  float* t0  = ws;
  float* h0  = ws + U;       // 1 unit
  float* h1  = ws + 2 * U;   // 3 units
  float* h2  = ws + 5 * U;   // 5 units
  float* yj0 = ws + 10 * U;
  float* yj1 = ws + 11 * U;
  float* yj2 = ws + 14 * U;
  float* v0  = ws + 19 * U;
  float* v1  = ws + 20 * U;
  float* v2  = ws + 23 * U;  // fp32 region ends at 28 units (~59 MB)
  float* out = (float*)d_out;
  const int N = NNODE;

  // ---- one-time fp16 conversion of all matrix weights -----------------
  _Float16* WH = (_Float16*)(ws + 28 * U);         // 27 x 128x128 f16 panels
  _Float16* MH = WH + 27 * (size_t)(FDIM * FDIM);  // 15 x 128x32  f16 panels
  static const int gemmW[27] = {
      82, 83, 84, 86, 87, 88, 28, 29, 30,   // branch j: lin1, lin2, linear_j
      70, 71, 72, 74, 75, 76, 24, 25, 26,   // branch i: lin1, lin2, linear_i
      58, 59, 60, 62, 63, 64, 32, 33, 34};  // post:     lin1, lin2, linear_v
  for (int k = 0; k < 27; ++k)
    cvt_f16_kernel<<<(FDIM * FDIM) / 256, 256, 0, stream>>>(
        pf(gemmW[k]), WH + (size_t)k * FDIM * FDIM, FDIM * FDIM);
  for (int p = 0; p < 15; ++p)
    cvt_f16_kernel<<<(FDIM * NBAS) / 256, 256, 0, stream>>>(
        pf(35 + p), MH + (size_t)p * FDIM * NBAS, FDIM * NBAS);
  auto wh = [&](int k) { return (const _Float16*)(WH + (size_t)k * FDIM * FDIM); };

  // ---------------- branch j: res_pre_vj -> swish(act_j) -> linear_j --------
  launch_swish(stream, x0, pf(77), pf(78), t0, N);
  launch_gemm(stream, t0, wh(0), pf(81), nullptr, h0, N, 1, 0, FDIM);
  launch_gemm(stream, x1, wh(1), nullptr, nullptr, h1, 3 * N, 1, 0, FDIM);
  launch_gemm(stream, x2, wh(2), nullptr, nullptr, h2, 5 * N, 1, 0, FDIM);
  launch_swish(stream, h0, pf(79), pf(80), t0, N);
  launch_gemm(stream, t0, wh(3), pf(85), x0, yj0, N, 1, 0, FDIM);
  launch_gemm(stream, h1, wh(4), nullptr, x1, yj1, 3 * N, 1, 0, FDIM);
  launch_gemm(stream, h2, wh(5), nullptr, x2, yj2, 5 * N, 1, 0, FDIM);
  launch_swish(stream, yj0, pf(11), pf(12), t0, N);
  launch_gemm(stream, t0, wh(6), pf(27), nullptr, yj0, N, 1, 0, FDIM);
  launch_gemm(stream, yj1, wh(7), nullptr, nullptr, yj1, 3 * N, 1, 0, FDIM);
  launch_gemm(stream, yj2, wh(8), nullptr, nullptr, yj2, 5 * N, 1, 0, FDIM);

  // ---------------- branch i: res_pre_vi -> swish(act_i) -> linear_i -> v ---
  launch_swish(stream, x0, pf(65), pf(66), t0, N);
  launch_gemm(stream, t0, wh(9), pf(69), nullptr, h0, N, 1, 0, FDIM);
  launch_gemm(stream, x1, wh(10), nullptr, nullptr, h1, 3 * N, 1, 0, FDIM);
  launch_gemm(stream, x2, wh(11), nullptr, nullptr, h2, 5 * N, 1, 0, FDIM);
  launch_swish(stream, h0, pf(67), pf(68), t0, N);
  launch_gemm(stream, t0, wh(12), pf(73), x0, v0, N, 1, 0, FDIM);
  launch_gemm(stream, h1, wh(13), nullptr, x1, v1, 3 * N, 1, 0, FDIM);
  launch_gemm(stream, h2, wh(14), nullptr, x2, v2, 5 * N, 1, 0, FDIM);
  launch_swish(stream, v0, pf(9), pf(10), t0, N);
  launch_gemm(stream, t0, wh(15), pf(23), nullptr, v0, N, 1, 0, FDIM);
  launch_gemm(stream, v1, wh(16), nullptr, nullptr, v1, 3 * N, 1, 0, FDIM);
  launch_gemm(stream, v2, wh(17), nullptr, nullptr, v2, 5 * N, 1, 0, FDIM);

  // ---------------- edge kernel: CG mixing + radial + scatter into v --------
  EdgeParams P;
  P.rbf = rbf; P.sph0 = sph0; P.sph1 = sph1; P.sph2 = sph2;
  P.idx_i = idx_i; P.idx_j = idx_j;
  P.yj0 = yj0; P.yj1 = yj1; P.yj2 = yj2;
  for (int p = 0; p < 15; ++p) P.mixh[p] = MH + (size_t)p * FDIM * NBAS;
  for (int L = 0; L < 3; ++L)  P.radw[L] = pf(50 + L);
  P.a1b = pf(15); P.a1w0 = pf(16); P.a1w1 = pf(17); P.a1w2 = pf(18);
  P.a2b = pf(19); P.a2w0 = pf(20); P.a2w1 = pf(21); P.a2w2 = pf(22);
  P.v0 = v0; P.v1 = v1; P.v2 = v2;
  edge_kernel<<<NEDGE / 16, 128, 0, stream>>>(P);

  // ---------------- res_post_v -> swish(act_v) -> linear_v -> out (+x) ------
  launch_swish(stream, v0, pf(53), pf(54), t0, N);
  launch_gemm(stream, t0, wh(18), pf(57), nullptr, h0, N, 1, 0, FDIM);
  launch_gemm(stream, v1, wh(19), nullptr, nullptr, h1, 3 * N, 1, 0, FDIM);
  launch_gemm(stream, v2, wh(20), nullptr, nullptr, h2, 5 * N, 1, 0, FDIM);
  launch_swish(stream, h0, pf(55), pf(56), t0, N);
  launch_gemm(stream, t0, wh(21), pf(61), v0, v0, N, 1, 0, FDIM);
  launch_gemm(stream, h1, wh(22), nullptr, v1, v1, 3 * N, 1, 0, FDIM);
  launch_gemm(stream, h2, wh(23), nullptr, v2, v2, 5 * N, 1, 0, FDIM);
  launch_swish(stream, v0, pf(13), pf(14), t0, N);
  // final linears fold residual (+x) and channel-concat layout [N, 9, 128]
  launch_gemm(stream, t0, wh(24), pf(31), x0, out, N,     1, 0, 9 * FDIM);
  launch_gemm(stream, v1, wh(25), nullptr, x1, out, 3 * N, 3, 1, 9 * FDIM);
  launch_gemm(stream, v2, wh(26), nullptr, x2, out, 5 * N, 5, 4, 9 * FDIM);
}
